// xFormersFill_vLLMPagedAttentionGenerate_31061203484871
// MI455X (gfx1250) — matrix operand order repaired and checked by reading, not compiled, not run on hardware
//
#include <hip/hip_runtime.h>
#include <hip/hip_bf16.h>

typedef __attribute__((ext_vector_type(16))) _Float16 v16h;
typedef __attribute__((ext_vector_type(8)))  float    v8f;

#define ATTN_SCALE 0.08838834764831845f   // 1/sqrt(128)
#define NHEAD 32
#define HDIM  128
#define NFILL 2048
#define FLEN  512
#define NGEN  16
#define BPS   64      // blocks per gen seq
#define GCTX  1024

// ---------- lane->element maps used here (A/B: contiguous k order, not the ISA table order) ----
// A-matrix 16x32 f16: lane holds row M=lane%16; element i -> K index:
__device__ __forceinline__ int a_kmap(int lane, int i) {
  int base = (lane < 16) ? 0 : 16;
  return base + i;
}
// B-matrix 32x16 f16: lane holds column N=lane%16; element i -> K = (lane<16?0:16)+i
// C/D 16x16 f32: lane holds column N=lane%16; vgpr j -> row = j + 8*(lane>=16)

__device__ __forceinline__ v8f wmma_f16(v16h a, v16h b, v8f c) {
  return __builtin_amdgcn_wmma_f32_16x16x32_f16(false, a, false, b, (short)0, c,
                                                false, false);
}

// =======================  FILL: flash attention  ============================
// One wave per (seq, 32-query tile, head); two 16-row WMMA subtiles per wave
// so K fragments, V staging and V fragments are amortized over 2x the WMMAs.
// 4 waves / 128-thread block.
__global__ __launch_bounds__(128) void fill_attn_kernel(
    const float* __restrict__ q, const float* __restrict__ k,
    const float* __restrict__ v, float* __restrict__ out) {
  __shared__ _Float16 Vt[4][32][132];      // V key-block tile, natural [key][d]
  __shared__ _Float16 Pb[4][2][16][34];    // P bounce buffer per subtile

  const int lane = threadIdx.x & 31;
  const int wv   = threadIdx.x >> 5;
  const int w    = blockIdx.x * 4 + wv;          // 0..2047
  const int head = w & 31;
  const int qt   = (w >> 5) & 15;                // 32-row q tile within seq
  const int seq  = w >> 9;                       // 0..3
  const int r16  = lane & 15;
  const int hi   = lane >> 4;                    // which half of wave

  // ---- Q fragments (A-layout) for both subtiles, 4 d-chunks of 32 ----
  v16h qa[2][4];
  for (int sub = 0; sub < 2; ++sub) {
    const int qtok = seq * FLEN + qt * 32 + sub * 16 + r16;
    const float* qp = q + ((size_t)qtok * NHEAD + head) * HDIM;
    for (int f = 0; f < 4; ++f)
      for (int i = 0; i < 16; ++i)
        qa[sub][f][i] = (_Float16)qp[f * 32 + a_kmap(lane, i)];
  }

  v8f acc[2][8];
  for (int s = 0; s < 2; ++s)
    for (int c = 0; c < 8; ++c)
      for (int j = 0; j < 8; ++j) acc[s][c][j] = 0.0f;
  float mrow[2][8], lrow[2][8];
  for (int s = 0; s < 2; ++s)
    for (int j = 0; j < 8; ++j) { mrow[s][j] = -1e30f; lrow[s][j] = 0.0f; }

  const int nkb = qt + 1;                        // causal: 32-key blocks needed
  for (int kb = 0; kb < nkb; ++kb) {
    const int kbase = kb * 32;

    // ---- stage V tile (f32 -> f16) into LDS, coalesced float4 rows ----
    for (int r = 0; r < 32; ++r) {
      const float* vp =
          v + ((size_t)(seq * FLEN + kbase + r) * NHEAD + head) * HDIM + lane * 4;
      float4 f = *(const float4*)vp;
      Vt[wv][r][lane * 4 + 0] = (_Float16)f.x;
      Vt[wv][r][lane * 4 + 1] = (_Float16)f.y;
      Vt[wv][r][lane * 4 + 2] = (_Float16)f.z;
      Vt[wv][r][lane * 4 + 3] = (_Float16)f.w;
    }
    if (kb + 1 < nkb)   // prefetch next K tile -> global_prefetch_b8
      __builtin_prefetch(
          k + ((size_t)(seq * FLEN + kbase + 32 + r16) * NHEAD + head) * HDIM, 0, 1);

    // ---- S = Q * K^T : K fragments built once, used by both subtiles ----
    v8f sc[2][2];   // [subtile][key tile]
    for (int s = 0; s < 2; ++s)
      for (int n = 0; n < 2; ++n)
        for (int j = 0; j < 8; ++j) sc[s][n][j] = 0.0f;
    for (int dc = 0; dc < 4; ++dc) {
      const int dof = dc * 32 + (hi ? 16 : 0);
      const float* kp0 =
          k + ((size_t)(seq * FLEN + kbase + r16) * NHEAD + head) * HDIM + dof;
      const float* kp1 =
          k + ((size_t)(seq * FLEN + kbase + 16 + r16) * NHEAD + head) * HDIM + dof;
      v16h b0, b1;
      for (int i = 0; i < 16; ++i) {
        b0[i] = (_Float16)kp0[i];
        b1[i] = (_Float16)kp1[i];
      }
      sc[0][0] = wmma_f16(qa[0][dc], b0, sc[0][0]);
      sc[0][1] = wmma_f16(qa[0][dc], b1, sc[0][1]);
      sc[1][0] = wmma_f16(qa[1][dc], b0, sc[1][0]);
      sc[1][1] = wmma_f16(qa[1][dc], b1, sc[1][1]);
    }

    // ---- scale + causal mask + online softmax, per subtile ----
    for (int sub = 0; sub < 2; ++sub) {
      float p0[8], p1[8], facs[8];
      for (int j = 0; j < 8; ++j) {
        const int qpos = qt * 32 + sub * 16 + j + 8 * hi;
        const int kp0pos = kbase + r16;
        const int kp1pos = kbase + 16 + r16;
        float x0 = (kp0pos <= qpos) ? sc[sub][0][j] * ATTN_SCALE : -1e30f;
        float x1 = (kp1pos <= qpos) ? sc[sub][1][j] * ATTN_SCALE : -1e30f;
        float rm = fmaxf(x0, x1);
        rm = fmaxf(rm, __shfl_xor(rm, 1, 32));
        rm = fmaxf(rm, __shfl_xor(rm, 2, 32));
        rm = fmaxf(rm, __shfl_xor(rm, 4, 32));
        rm = fmaxf(rm, __shfl_xor(rm, 8, 32));
        float mn  = fmaxf(mrow[sub][j], rm);
        float fac = __expf(mrow[sub][j] - mn);
        float e0  = __expf(x0 - mn);
        float e1  = __expf(x1 - mn);
        float rs  = e0 + e1;
        rs += __shfl_xor(rs, 1, 32);
        rs += __shfl_xor(rs, 2, 32);
        rs += __shfl_xor(rs, 4, 32);
        rs += __shfl_xor(rs, 8, 32);
        lrow[sub][j] = lrow[sub][j] * fac + rs;
        mrow[sub][j] = mn;
        facs[j] = fac;
        p0[j] = e0;
        p1[j] = e1;
      }
      for (int c = 0; c < 8; ++c)
        for (int j = 0; j < 8; ++j) acc[sub][c][j] *= facs[j];
      // repack P: C-layout -> A-layout via LDS (same wave, in-order)
      for (int j = 0; j < 8; ++j) {
        Pb[wv][sub][j + 8 * hi][r16]      = (_Float16)p0[j];
        Pb[wv][sub][j + 8 * hi][16 + r16] = (_Float16)p1[j];
      }
    }
    v16h pa0, pa1;
    for (int i = 0; i < 16; ++i) {
      const int km = a_kmap(lane, i);
      pa0[i] = Pb[wv][0][r16][km];
      pa1[i] = Pb[wv][1][r16][km];
    }

    // ---- O += P * V : V fragments built once, used by both subtiles ----
    const int kb2 = hi ? 16 : 0;
    for (int c = 0; c < 8; ++c) {
      v16h vb;
      for (int i = 0; i < 16; ++i) vb[i] = Vt[wv][kb2 + i][c * 16 + r16];
      acc[0][c] = wmma_f16(pa0, vb, acc[0][c]);
      acc[1][c] = wmma_f16(pa1, vb, acc[1][c]);
    }
  }

  // ---- epilogue: normalize and store (C-layout scatter) ----
  for (int sub = 0; sub < 2; ++sub)
    for (int c = 0; c < 8; ++c)
      for (int j = 0; j < 8; ++j) {
        const int row = j + 8 * hi;
        const size_t idx =
            ((size_t)(seq * FLEN + qt * 32 + sub * 16 + row)) * (NHEAD * HDIM) +
            head * HDIM + c * 16 + r16;
        out[idx] = acc[sub][c][j] / lrow[sub][j];
      }
}

// =====================  GEN: paged decode attention  ========================
// One 256-thread block per (gen seq, head). Bandwidth-bound -> VALU.
__global__ __launch_bounds__(256) void gen_attn_kernel(
    const float* __restrict__ q, const float* __restrict__ k,
    const float* __restrict__ v, const float* __restrict__ kc,
    const float* __restrict__ vc, const int* __restrict__ slot_map,
    const int* __restrict__ bt, const int* __restrict__ ctx_lens,
    float* __restrict__ out) {
  __shared__ float sc[GCTX];
  __shared__ float red[256];
  __shared__ float obuf[256];

  const int g    = blockIdx.x >> 5;
  const int head = blockIdx.x & 31;
  const int t    = threadIdx.x;
  const int lane = t & 31;
  const int wv   = t >> 5;
  const int ctx  = ctx_lens[g];
  const int sub_slot = slot_map[NFILL + g];   // where this gen token's k/v lives
  const size_t newtok = ((size_t)(NFILL + g) * NHEAD + head) * HDIM;

  float qr[4];
  for (int j = 0; j < 4; ++j) qr[j] = q[newtok + lane + 32 * j];

  // scores: each wave owns 128 keys, lane-over-d dot products
  for (int r = 0; r < 128; ++r) {
    const int i = wv * 128 + r;
    float s;
    if (i < ctx) {
      const int blk  = bt[g * BPS + (i >> 4)];
      const int slot = blk * 16 + (i & 15);
      const float* kp = (slot == sub_slot) ? (k + newtok)
                                           : (kc + ((size_t)slot * NHEAD + head) * HDIM);
      float p = 0.0f;
      for (int j = 0; j < 4; ++j) p += qr[j] * kp[lane + 32 * j];
      p += __shfl_xor(p, 16, 32);
      p += __shfl_xor(p, 8, 32);
      p += __shfl_xor(p, 4, 32);
      p += __shfl_xor(p, 2, 32);
      p += __shfl_xor(p, 1, 32);
      s = p * ATTN_SCALE;
    } else {
      s = -1e30f;
    }
    if (lane == 0) sc[i] = s;
  }
  __syncthreads();

  // block softmax over 1024 scores
  float lm = -1e30f;
  for (int i = t; i < GCTX; i += 256) lm = fmaxf(lm, sc[i]);
  red[t] = lm;
  __syncthreads();
  for (int s = 128; s > 0; s >>= 1) {
    if (t < s) red[t] = fmaxf(red[t], red[t + s]);
    __syncthreads();
  }
  const float m = red[0];
  __syncthreads();

  float ls = 0.0f;
  for (int i = t; i < GCTX; i += 256) {
    float e = __expf(sc[i] - m);
    sc[i] = e;
    ls += e;
  }
  red[t] = ls;
  __syncthreads();
  for (int s = 128; s > 0; s >>= 1) {
    if (t < s) red[t] += red[t + s];
    __syncthreads();
  }
  const float l = red[0];
  __syncthreads();

  // P*V: thread owns d = t%128 over half the keys; combine halves in LDS
  const int d = t & 127;
  const int halfsel = t >> 7;
  float a = 0.0f;
  const int i0 = halfsel * 512;
  const int i1 = min(i0 + 512, ctx);
  for (int i = i0; i < i1; ++i) {
    const int blk  = bt[g * BPS + (i >> 4)];
    const int slot = blk * 16 + (i & 15);
    const float* vp = (slot == sub_slot) ? (v + newtok)
                                         : (vc + ((size_t)slot * NHEAD + head) * HDIM);
    a += sc[i] * vp[d];
  }
  obuf[t] = a;
  __syncthreads();
  if (t < 128)
    out[((size_t)(NFILL + g)) * (NHEAD * HDIM) + head * HDIM + t] =
        (obuf[t] + obuf[t + 128]) / l;
}

extern "C" void kernel_launch(void* const* d_in, const int* in_sizes, int n_in,
                              void* d_out, int out_size, void* d_ws, size_t ws_size,
                              hipStream_t stream) {
  const float* q  = (const float*)d_in[0];
  const float* k  = (const float*)d_in[1];
  const float* v  = (const float*)d_in[2];
  const float* kc = (const float*)d_in[3];
  const float* vc = (const float*)d_in[4];
  const int* slot_map = (const int*)d_in[5];
  const int* bt       = (const int*)d_in[6];
  const int* ctx      = (const int*)d_in[7];
  float* out = (float*)d_out;

  // fill: 4 seqs * 16 q-tiles(32 rows) * 32 heads = 2048 waves / 4 per block
  fill_attn_kernel<<<512, 128, 0, stream>>>(q, k, v, out);
  // gen: 16 seqs * 32 heads
  gen_attn_kernel<<<NGEN * NHEAD, 256, 0, stream>>>(q, k, v, kc, vc, slot_map,
                                                    bt, ctx, out);
}
